// MultiQueryAttention_10213432229946
// MI455X (gfx1250) — matrix operand order repaired and checked by reading, not compiled, not run on hardware
//
#include <hip/hip_runtime.h>
#include <hip/hip_bf16.h>
#include <stdint.h>

// ---------------------------------------------------------------------------
// Types for CDNA5 WMMA (wave32): bf16 A/B fragments (16 bf16 = 8 VGPRs),
// fp32 C/D accumulators (8 floats = 8 VGPRs).
// ---------------------------------------------------------------------------
typedef __attribute__((ext_vector_type(16))) __bf16 v16bf;
typedef __attribute__((ext_vector_type(8)))  float  v8f;

union Frag16 {
    v16bf v;
    uint4 q[2];
};

__device__ __forceinline__ unsigned short f2bf(float f) {
    unsigned u = __builtin_bit_cast(unsigned, f);
    u += 0x7fffu + ((u >> 16) & 1u);   // round-to-nearest-even
    return (unsigned short)(u >> 16);
}

// Async global -> LDS copy (16B per lane), tracked by ASYNCcnt.
// VDST operand is a per-lane LDS byte address; low 32 bits of a flat
// shared-memory pointer are exactly the LDS offset (ISA 10.2 aperture rules).
__device__ __forceinline__ void async_ld_b128(unsigned lds_addr, const void* g) {
    asm volatile("global_load_async_to_lds_b128 %0, %1, off"
                 :: "v"(lds_addr), "v"(g)
                 : "memory");
}
__device__ __forceinline__ void wait_async0() {
    asm volatile("s_wait_asynccnt 0x0" ::: "memory");
}

// ---------------------------------------------------------------------------
// fp32 -> bf16 elementwise convert (vectorized 4-wide)
// ---------------------------------------------------------------------------
__global__ void cvt_bf16(const float* __restrict__ in,
                         unsigned short* __restrict__ out, int n4) {
    int i = blockIdx.x * blockDim.x + threadIdx.x;
    for (; i < n4; i += gridDim.x * blockDim.x) {
        float4 f = ((const float4*)in)[i];
        ushort4 r;
        r.x = f2bf(f.x); r.y = f2bf(f.y); r.z = f2bf(f.z); r.w = f2bf(f.w);
        ((ushort4*)out)[i] = r;
    }
}

// ---------------------------------------------------------------------------
// Transpose + convert: in[K][N] fp32 -> out[N][K] bf16, 32x32 LDS tiles.
// ---------------------------------------------------------------------------
__global__ __launch_bounds__(256)
void transpose_cvt(const float* __restrict__ in,
                   unsigned short* __restrict__ out, int K, int N) {
    __shared__ float tile[32][33];
    const int k0 = blockIdx.y * 32, n0 = blockIdx.x * 32;
    const int tx = threadIdx.x & 31, ty = threadIdx.x >> 5;
    #pragma unroll
    for (int i = 0; i < 32; i += 8)
        tile[ty + i][tx] = in[(size_t)(k0 + ty + i) * N + n0 + tx];
    __syncthreads();
    #pragma unroll
    for (int i = 0; i < 32; i += 8)
        out[(size_t)(n0 + ty + i) * K + k0 + tx] = f2bf(tile[tx][ty + i]);
}

// ---------------------------------------------------------------------------
// bf16 WMMA GEMM: C[M][N] = A[M][K] @ Bt[N][K]^T, fp32 accumulate.
// Block tile 128x128, 8 waves 4(M)x2(N), each wave 32x64, K stepped by 32.
// OUT_MODE 0: fp32 out + bias; 1: bf16 out, (acc+bias)*scale;
//          2: bf16 out transposed -> out[n*M + m]  (builds V^T)
// ---------------------------------------------------------------------------
template <int OUT_MODE>
__global__ __launch_bounds__(256)
void gemm_bf16(const unsigned short* __restrict__ A,
               const unsigned short* __restrict__ Bt,
               const float* __restrict__ bias,
               void* __restrict__ Cout,
               int M, int N, int K, float outScale) {
    const int lane  = threadIdx.x & 31;
    const int wave  = threadIdx.x >> 5;
    const int l15   = lane & 15;
    const int half  = lane >> 4;
    const int mBase = blockIdx.y * 128 + (wave & 3) * 32;
    const int nBase = blockIdx.x * 128 + (wave >> 2) * 64;

    v8f acc[2][4];
    #pragma unroll
    for (int mt = 0; mt < 2; ++mt)
        #pragma unroll
        for (int nt = 0; nt < 4; ++nt)
            acc[mt][nt] = (v8f){0, 0, 0, 0, 0, 0, 0, 0};

    for (int k0 = 0; k0 < K; k0 += 32) {
        // Line-ahead prefetch (global_prefetch_b8) once per 4 k-steps.
        if ((k0 & 127) == 0 && k0 + 160 < K) {
            #pragma unroll
            for (int mt = 0; mt < 2; ++mt)
                __builtin_prefetch(A + (size_t)(mBase + mt * 16 + l15) * K + k0 + 128, 0, 3);
            #pragma unroll
            for (int nt = 0; nt < 4; ++nt)
                __builtin_prefetch(Bt + (size_t)(nBase + nt * 16 + l15) * K + k0 + 128, 0, 3);
        }
        Frag16 a[2], b[4];
        #pragma unroll
        for (int mt = 0; mt < 2; ++mt) {
            const unsigned short* p =
                A + (size_t)(mBase + mt * 16 + l15) * K + k0 + half * 8;
            a[mt].q[0] = *(const uint4*)p;
            a[mt].q[1] = *(const uint4*)(p + 16);
        }
        #pragma unroll
        for (int nt = 0; nt < 4; ++nt) {
            const unsigned short* p =
                Bt + (size_t)(nBase + nt * 16 + l15) * K + k0 + half * 8;
            b[nt].q[0] = *(const uint4*)p;
            b[nt].q[1] = *(const uint4*)(p + 16);
        }
        #pragma unroll
        for (int mt = 0; mt < 2; ++mt)
            #pragma unroll
            for (int nt = 0; nt < 4; ++nt)
                acc[mt][nt] = __builtin_amdgcn_wmma_f32_16x16x32_bf16(
                    false, a[mt].v, false, b[nt].v, (short)0, acc[mt][nt],
                    false, false);
    }

    #pragma unroll
    for (int nt = 0; nt < 4; ++nt) {
        const int n = nBase + nt * 16 + l15;
        const float bv = bias[n];
        #pragma unroll
        for (int mt = 0; mt < 2; ++mt) {
            #pragma unroll
            for (int r = 0; r < 8; ++r) {
                const int m = mBase + mt * 16 + half * 8 + r;
                const float val = (acc[mt][nt][r] + bv) * outScale;
                if (OUT_MODE == 0)
                    ((float*)Cout)[(size_t)m * N + n] = val;
                else if (OUT_MODE == 1)
                    ((unsigned short*)Cout)[(size_t)m * N + n] = f2bf(val);
                else
                    ((unsigned short*)Cout)[(size_t)n * M + m] = f2bf(val);
            }
        }
    }
}

// ---------------------------------------------------------------------------
// Flash-attention MQA core with async double-buffered K/V staging.
// Grid: (S/128, B*H). Block: 256 threads = 8 waves; wave w owns 16 q-rows.
// Qb [B*S][H*Dh] bf16 (pre-scaled); Kb [B*S][Dh]; Vt [Dh][B*S]; Ctx out bf16.
// LDS: 2x32KB K buffers (current one reused for P tiles after barrier)
//    + 2x32KB V^T buffers = 128KB.
// ---------------------------------------------------------------------------
__global__ __launch_bounds__(256)
void mqa_attention(const unsigned short* __restrict__ Qb,
                   const unsigned short* __restrict__ Kb,
                   const unsigned short* __restrict__ Vt,
                   unsigned short* __restrict__ Ctx) {
    constexpr int S = 2048, Dh = 128, KB = 128, Dm = 2048, Mtot = 4096;
    __shared__ unsigned short bufK[2][KB * Dh];
    __shared__ unsigned short bufV[2][Dh * KB];

    const int lane = threadIdx.x & 31;
    const int wave = threadIdx.x >> 5;
    const int l15  = lane & 15;
    const int half = lane >> 4;

    const int b = (int)blockIdx.y >> 4;
    const int h = (int)blockIdx.y & 15;
    const int qRow0 = blockIdx.x * 128 + wave * 16;

    // Issue async staging for one 128-key block into buffer `ib`.
    auto stage = [&](int kb, int ib) {
        unsigned dk = (unsigned)(size_t)&bufK[ib][0];
        const unsigned short* srcK = Kb + ((size_t)b * S + kb) * Dh;  // contiguous
        for (int i = threadIdx.x; i < KB * Dh / 8; i += 256)
            async_ld_b128(dk + i * 16, srcK + i * 8);
        unsigned dv = (unsigned)(size_t)&bufV[ib][0];
        for (int i = threadIdx.x; i < Dh * KB / 8; i += 256) {
            const int d = i >> 4, c = i & 15;   // 16 x b128 per V^T row
            async_ld_b128(dv + i * 16, Vt + (size_t)d * Mtot + b * S + kb + c * 8);
        }
    };

    // Q fragments: 4 x (16x32) over head dim, loaded once
    Frag16 qf[4];
    #pragma unroll
    for (int ds = 0; ds < 4; ++ds) {
        const unsigned short* p = Qb + (size_t)(b * S + qRow0 + l15) * Dm +
                                  h * Dh + ds * 32 + half * 8;
        qf[ds].q[0] = *(const uint4*)p;
        qf[ds].q[1] = *(const uint4*)(p + 16);
    }

    v8f outAcc[8];
    #pragma unroll
    for (int dt = 0; dt < 8; ++dt) outAcc[dt] = (v8f){0, 0, 0, 0, 0, 0, 0, 0};
    float mRun[8], lRun[8];
    #pragma unroll
    for (int r = 0; r < 8; ++r) { mRun[r] = -3.0e38f; lRun[r] = 0.0f; }

    stage(0, 0);   // prologue prefetch of block 0

    int ib = 0;
    for (int kb = 0; kb < S; kb += KB, ib ^= 1) {
        wait_async0();       // current block resident in bufK/bufV[ib]
        __syncthreads();     // and everyone is done with buffer ib^1
        if (kb + KB < S) stage(kb + KB, ib ^ 1);   // prefetch next block

        // scores: 16 q-rows x 128 keys = 8 tiles x 4 K-steps = 32 WMMAs
        v8f s[8];
        #pragma unroll
        for (int t = 0; t < 8; ++t) s[t] = (v8f){0, 0, 0, 0, 0, 0, 0, 0};
        #pragma unroll
        for (int t = 0; t < 8; ++t) {
            #pragma unroll
            for (int ds = 0; ds < 4; ++ds) {
                Frag16 bf;
                const unsigned short* p =
                    &bufK[ib][(t * 16 + l15) * Dh + ds * 32 + half * 8];
                bf.q[0] = *(const uint4*)p;
                bf.q[1] = *(const uint4*)(p + 16);
                s[t] = __builtin_amdgcn_wmma_f32_16x16x32_bf16(
                    false, qf[ds].v, false, bf.v, (short)0, s[t], false, false);
            }
        }
        __syncthreads();   // all waves done reading K block -> reuse for P

        // Online softmax. C layout: row m = half*8+r spans one 16-lane half.
        float mNew[8], corr[8];
        #pragma unroll
        for (int r = 0; r < 8; ++r) {
            float mx = s[0][r];
            #pragma unroll
            for (int t = 1; t < 8; ++t) mx = fmaxf(mx, s[t][r]);
            #pragma unroll
            for (int msk = 1; msk < 16; msk <<= 1)
                mx = fmaxf(mx, __shfl_xor(mx, msk, 32));
            mNew[r] = fmaxf(mRun[r], mx);
            corr[r] = __expf(mRun[r] - mNew[r]);
            mRun[r] = mNew[r];
            lRun[r] *= corr[r];
        }
        #pragma unroll
        for (int dt = 0; dt < 8; ++dt)
            #pragma unroll
            for (int r = 0; r < 8; ++r) outAcc[dt][r] *= corr[r];

        // P = exp(s - mNew): row sums + bf16 tile to per-wave LDS region
        unsigned short* pW = &bufK[ib][wave * (16 * KB)];
        float rs[8];
        #pragma unroll
        for (int r = 0; r < 8; ++r) rs[r] = 0.0f;
        #pragma unroll
        for (int t = 0; t < 8; ++t) {
            #pragma unroll
            for (int r = 0; r < 8; ++r) {
                const float pv = __expf(s[t][r] - mNew[r]);
                rs[r] += pv;
                pW[(half * 8 + r) * KB + t * 16 + l15] = f2bf(pv);
            }
        }
        #pragma unroll
        for (int r = 0; r < 8; ++r) {
            float v = rs[r];
            #pragma unroll
            for (int msk = 1; msk < 16; msk <<= 1) v += __shfl_xor(v, msk, 32);
            lRun[r] += v;
        }

        // Reload P as A fragments (C->A layout fixup via LDS), then PV
        Frag16 pf[4];
        #pragma unroll
        for (int ks = 0; ks < 4; ++ks) {
            const unsigned short* p = pW + l15 * KB + ks * 32 + half * 8;
            pf[ks].q[0] = *(const uint4*)p;
            pf[ks].q[1] = *(const uint4*)(p + 16);
        }
        #pragma unroll
        for (int dt = 0; dt < 8; ++dt) {
            #pragma unroll
            for (int ks = 0; ks < 4; ++ks) {
                Frag16 vf;
                const unsigned short* p =
                    &bufV[ib][(dt * 16 + l15) * KB + ks * 32 + half * 8];
                vf.q[0] = *(const uint4*)p;
                vf.q[1] = *(const uint4*)(p + 16);
                outAcc[dt] = __builtin_amdgcn_wmma_f32_16x16x32_bf16(
                    false, pf[ks].v, false, vf.v, (short)0, outAcc[dt],
                    false, false);
            }
        }
    }

    #pragma unroll
    for (int r = 0; r < 8; ++r) lRun[r] = 1.0f / lRun[r];
    #pragma unroll
    for (int dt = 0; dt < 8; ++dt)
        #pragma unroll
        for (int r = 0; r < 8; ++r) {
            const float v = outAcc[dt][r] * lRun[r];
            Ctx[(size_t)(b * S + qRow0 + half * 8 + r) * Dm + h * Dh +
                dt * 16 + l15] = f2bf(v);
        }
}

// ---------------------------------------------------------------------------
// Launch pipeline
// ---------------------------------------------------------------------------
extern "C" void kernel_launch(void* const* d_in, const int* in_sizes, int n_in,
                              void* d_out, int out_size, void* d_ws,
                              size_t ws_size, hipStream_t stream) {
    const float* x  = (const float*)d_in[0];
    const float* wq = (const float*)d_in[1];
    const float* bq = (const float*)d_in[2];
    const float* wk = (const float*)d_in[3];
    const float* bk = (const float*)d_in[4];
    const float* wv = (const float*)d_in[5];
    const float* bv = (const float*)d_in[6];
    const float* wo = (const float*)d_in[7];
    const float* bo = (const float*)d_in[8];
    float* out = (float*)d_out;

    constexpr int Btch = 2, S = 2048, D = 2048, Dh = 128;
    constexpr int M = Btch * S;   // 4096

    unsigned short* ws = (unsigned short*)d_ws;
    size_t o = 0;
    unsigned short* xb  = ws + o; o += (size_t)M * D;    // x  bf16 [M][D]
    unsigned short* wqT = ws + o; o += (size_t)D * D;    // wq^T [D][D]
    unsigned short* wkT = ws + o; o += (size_t)Dh * D;   // wk^T [Dh][D]
    unsigned short* wvT = ws + o; o += (size_t)Dh * D;   // wv^T [Dh][D]
    unsigned short* woT = ws + o; o += (size_t)D * D;    // wo^T [D][D]
    unsigned short* qb  = ws + o; o += (size_t)M * D;    // Q*scale bf16
    unsigned short* kbf = ws + o; o += (size_t)M * Dh;   // K bf16 [M][Dh]
    unsigned short* vt  = ws + o; o += (size_t)Dh * M;   // V^T bf16 [Dh][M]
    unsigned short* ctx = ws + o; o += (size_t)M * D;    // attn out bf16

    cvt_bf16<<<1024, 256, 0, stream>>>(x, xb, M * D / 4);
    transpose_cvt<<<dim3(D / 32, D / 32), 256, 0, stream>>>(wq, wqT, D, D);
    transpose_cvt<<<dim3(Dh / 32, D / 32), 256, 0, stream>>>(wk, wkT, D, Dh);
    transpose_cvt<<<dim3(Dh / 32, D / 32), 256, 0, stream>>>(wv, wvT, D, Dh);
    transpose_cvt<<<dim3(D / 32, D / 32), 256, 0, stream>>>(wo, woT, D, D);

    const float qscale = 0.08838834764831845f;   // 1/sqrt(128)
    gemm_bf16<1><<<dim3(D / 128, M / 128), 256, 0, stream>>>(
        xb, wqT, bq, qb, M, D, D, qscale);
    gemm_bf16<1><<<dim3(1, M / 128), 256, 0, stream>>>(
        xb, wkT, bk, kbf, M, Dh, D, 1.0f);
    gemm_bf16<2><<<dim3(1, M / 128), 256, 0, stream>>>(
        xb, wvT, bv, vt, M, Dh, D, 1.0f);

    mqa_attention<<<dim3(S / 128, Btch * 16), 256, 0, stream>>>(qb, kbf, vt, ctx);

    gemm_bf16<0><<<dim3(D / 128, M / 128), 256, 0, stream>>>(
        ctx, woT, bo, out, M, D, D, 1.0f);
}
